// DebertaV2Attention_33621003993566
// MI455X (gfx1250) — compile-verified, matrix operand order checked by reading, not run on hardware
//
#include <hip/hip_runtime.h>
#include <hip/hip_bf16.h>

// ---------------------------------------------------------------------------
// DeBERTa-v2 disentangled attention for MI455X (gfx1250, wave32, WMMA).
// v_wmma_f32_16x16x32_f16 everywhere; all fragments 2 x b128 per lane;
// software-pipelined (double-buffered) K loops to hide load latency.
// ---------------------------------------------------------------------------

constexpr int Bc  = 4;
constexpr int Sc  = 1024;
constexpr int Hc  = 1536;
constexpr int NHc = 24;
constexpr int DHc = 64;
constexpr int PT  = 512;   // 2 * ATT_SPAN

typedef __attribute__((ext_vector_type(16))) _Float16 v16h;
typedef __attribute__((ext_vector_type(8)))  _Float16 v8h;
typedef __attribute__((ext_vector_type(4)))  _Float16 v4h;
typedef __attribute__((ext_vector_type(8)))  float    v8f;

union AF { v16h v; v8h q[2]; _Float16 h[16]; };
union CF { v8f  v; float f[8]; };

// Load a 16x32 f16 A/B fragment whose K dimension is contiguous at p0.
// ISA 7.12.2: per lane the fragment is two contiguous 8-half (16B) runs at
// K = k0..k0+7 and K = k0+16..k0+23 with k0 = 8*(lane>=16)  ->  2 x b128.
static __device__ __forceinline__ void load_frag(AF& a, const _Float16* p0,
                                                 int lane) {
  const _Float16* p = p0 + ((lane >> 4) << 3);
  a.q[0] = *(const v8h*)(p);
  a.q[1] = *(const v8h*)(p + 16);
}

static __device__ __forceinline__ void wmma_acc(const AF& a, const AF& b, CF& c) {
  c.v = __builtin_amdgcn_wmma_f32_16x16x32_f16(false, a.v, false, b.v,
                                               (short)0, c.v, false, false);
}

// ---------------------------------------------------------------------------
// Elementwise f32 -> f16.
// ---------------------------------------------------------------------------
__global__ __launch_bounds__(256) void k_cvt16(const float* __restrict__ x,
                                               _Float16* __restrict__ y, int n)
{
  const int i = (blockIdx.x * 256 + threadIdx.x) * 4;
  if (i + 3 < n) {
    const float4 v = *(const float4*)(x + i);
    v4h o;
    o.x = (_Float16)v.x; o.y = (_Float16)v.y;
    o.z = (_Float16)v.z; o.w = (_Float16)v.w;
    *(v4h*)(y + i) = o;
  }
}

// ---------------------------------------------------------------------------
// Wt_f16[N][K] = W_f32[K][N]^T  (LDS-tiled, coalesced both sides).
// ---------------------------------------------------------------------------
__global__ __launch_bounds__(128) void k_wT(const float* __restrict__ W,
                                            _Float16* __restrict__ Wt,
                                            int K, int N)
{
  __shared__ _Float16 tile[32][33];
  const int n0 = blockIdx.x * 32;
  const int k0 = blockIdx.y * 32;
  const int c  = threadIdx.x & 31;
  const int r0 = threadIdx.x >> 5;
  for (int r = r0; r < 32; r += 4)
    tile[r][c] = (_Float16)W[(size_t)(k0 + r) * N + n0 + c];
  __syncthreads();
  for (int r = r0; r < 32; r += 4)
    Wt[(size_t)(n0 + r) * K + k0 + c] = tile[c][r];
}

// ---------------------------------------------------------------------------
// Software-pipelined GEMM body shared by the three GEMM kernels.
// Wave tile 32x64; unroll-by-2 over K with two fragment register sets so the
// loads for step k+32 are in flight during the 8 WMMAs of step k.
// ---------------------------------------------------------------------------
#define GEMM_PIPELINE_BODY()                                                   \
  CF acc[2][4];                                                                \
  _Pragma("unroll") for (int s = 0; s < 2; ++s)                                \
  _Pragma("unroll") for (int t = 0; t < 4; ++t)                                \
  _Pragma("unroll") for (int i = 0; i < 8; ++i) acc[s][t].f[i] = 0.f;          \
  const _Float16* arow0 = A + (size_t)(m0 + lo) * K;                           \
  const _Float16* arow1 = A + (size_t)(m0 + 16 + lo) * K;                      \
  const _Float16* wrow0 = Wt + (size_t)(n0 + lo) * K;                          \
  const _Float16* wrow1 = Wt + (size_t)(n0 + 16 + lo) * K;                     \
  const _Float16* wrow2 = Wt + (size_t)(n0 + 32 + lo) * K;                     \
  const _Float16* wrow3 = Wt + (size_t)(n0 + 48 + lo) * K;                     \
  AF a0A, a1A, b0A, b1A, b2A, b3A;                                             \
  AF a0B, a1B, b0B, b1B, b2B, b3B;                                             \
  auto load_set = [&](AF& A0, AF& A1, AF& B0, AF& B1, AF& B2, AF& B3, int k) { \
    load_frag(A0, arow0 + k, lane);                                            \
    load_frag(A1, arow1 + k, lane);                                            \
    load_frag(B0, wrow0 + k, lane);                                            \
    load_frag(B1, wrow1 + k, lane);                                            \
    load_frag(B2, wrow2 + k, lane);                                            \
    load_frag(B3, wrow3 + k, lane);                                            \
  };                                                                           \
  auto mm8 = [&](AF& A0, AF& A1, AF& B0, AF& B1, AF& B2, AF& B3) {             \
    wmma_acc(A0, B0, acc[0][0]); wmma_acc(A1, B0, acc[1][0]);                  \
    wmma_acc(A0, B1, acc[0][1]); wmma_acc(A1, B1, acc[1][1]);                  \
    wmma_acc(A0, B2, acc[0][2]); wmma_acc(A1, B2, acc[1][2]);                  \
    wmma_acc(A0, B3, acc[0][3]); wmma_acc(A1, B3, acc[1][3]);                  \
  };                                                                           \
  load_set(a0A, a1A, b0A, b1A, b2A, b3A, 0);                                   \
  for (int k0 = 0; k0 < K; k0 += 64) {                                         \
    load_set(a0B, a1B, b0B, b1B, b2B, b3B, k0 + 32);                           \
    __builtin_prefetch(arow0 + k0 + 128, 0, 1);                                \
    __builtin_prefetch(wrow0 + k0 + 128, 0, 1);                                \
    mm8(a0A, a1A, b0A, b1A, b2A, b3A);                                         \
    if (k0 + 64 < K) load_set(a0A, a1A, b0A, b1A, b2A, b3A, k0 + 64);          \
    mm8(a0B, a1B, b0B, b1B, b2B, b3B);                                         \
  }

// ---------------------------------------------------------------------------
// GEMM: C_f16[M,N] = A_f16[M,K] @ Wt_f16[N,K]^T + bias
// ---------------------------------------------------------------------------
__global__ __launch_bounds__(128) void k_gemm16(
    const _Float16* __restrict__ A, const _Float16* __restrict__ Wt,
    const float* __restrict__ bias, _Float16* __restrict__ C,
    int M, int N, int K)
{
  const int lane = threadIdx.x & 31;
  const int wave = threadIdx.x >> 5;
  const int lo   = lane & 15;
  const int hi8  = (lane >> 4) << 3;
  const int m0   = blockIdx.y * 128 + wave * 32;
  const int n0   = blockIdx.x * 64;
  (void)M;

  GEMM_PIPELINE_BODY()

#pragma unroll
  for (int s = 0; s < 2; ++s)
#pragma unroll
    for (int t = 0; t < 4; ++t) {
      const int col = n0 + t * 16 + lo;
      const float bb = bias[col];
#pragma unroll
      for (int i = 0; i < 8; ++i) {
        const int row = m0 + s * 16 + i + hi8;
        C[(size_t)row * N + col] = (_Float16)(acc[s][t].f[i] + bb);
      }
    }
}

// ---------------------------------------------------------------------------
// Same GEMM, stores head-transposed output for V:
//   vt[((b*NH + h)*DH + d)*S + s]  so PV fragments are K-contiguous.
// ---------------------------------------------------------------------------
__global__ __launch_bounds__(128) void k_gemm16_vT(
    const _Float16* __restrict__ A, const _Float16* __restrict__ Wt,
    const float* __restrict__ bias, _Float16* __restrict__ vt,
    int M, int N, int K)
{
  const int lane = threadIdx.x & 31;
  const int wave = threadIdx.x >> 5;
  const int lo   = lane & 15;
  const int hi8  = (lane >> 4) << 3;
  const int m0   = blockIdx.y * 128 + wave * 32;
  const int n0   = blockIdx.x * 64;
  (void)M;

  GEMM_PIPELINE_BODY()

  const int head = n0 / DHc;
#pragma unroll
  for (int s = 0; s < 2; ++s)
#pragma unroll
    for (int t = 0; t < 4; ++t) {
      const int dloc = t * 16 + lo;
      const float bb = bias[n0 + dloc];
#pragma unroll
      for (int i = 0; i < 8; ++i) {
        const int row = m0 + s * 16 + i + hi8;     // = b*S + s_tok
        const int bidx = row / Sc;
        const int stok = row - bidx * Sc;
        vt[(((size_t)bidx * NHc + head) * DHc + dloc) * Sc + stok] =
            (_Float16)(acc[s][t].f[i] + bb);
      }
    }
}

// ---------------------------------------------------------------------------
// Output projection + residual: C_f32 = A_f16 @ Wt^T + bias + resid_f32
// ---------------------------------------------------------------------------
__global__ __launch_bounds__(128) void k_gemm16_out(
    const _Float16* __restrict__ A, const _Float16* __restrict__ Wt,
    const float* __restrict__ bias, const float* __restrict__ resid,
    float* __restrict__ C, int M, int N, int K)
{
  const int lane = threadIdx.x & 31;
  const int wave = threadIdx.x >> 5;
  const int lo   = lane & 15;
  const int hi8  = (lane >> 4) << 3;
  const int m0   = blockIdx.y * 128 + wave * 32;
  const int n0   = blockIdx.x * 64;
  (void)M;

  GEMM_PIPELINE_BODY()

#pragma unroll
  for (int s = 0; s < 2; ++s)
#pragma unroll
    for (int t = 0; t < 4; ++t) {
      const int col = n0 + t * 16 + lo;
      const float bb = bias[col];
#pragma unroll
      for (int i = 0; i < 8; ++i) {
        const int row = m0 + s * 16 + i + hi8;
        C[(size_t)row * N + col] =
            acc[s][t].f[i] + bb + resid[(size_t)row * N + col];
      }
    }
}

// ---------------------------------------------------------------------------
// p2c_full[bh][s][p] = sum_d k[b,s,h,d] * pos_q[p,h,d]   (f16 out)
// grid = (S/64, B*NH), block 128; double-buffered over p-tiles.
// ---------------------------------------------------------------------------
__global__ __launch_bounds__(128) void k_p2c(
    const _Float16* __restrict__ kh, const _Float16* __restrict__ posq,
    _Float16* __restrict__ p2c)
{
  const int lane = threadIdx.x & 31;
  const int wave = threadIdx.x >> 5;
  const int lo   = lane & 15;
  const int hi8  = (lane >> 4) << 3;
  const int bh   = blockIdx.y;
  const int bb   = bh / NHc;
  const int hh   = bh % NHc;
  const int r0   = blockIdx.x * 64 + wave * 16;

  const _Float16* kb = kh + (size_t)bb * Sc * Hc + hh * DHc;
  const _Float16* pb = posq + hh * DHc;

  AF a0, a1;
  {
    const _Float16* ka = kb + (size_t)(r0 + lo) * Hc;
    load_frag(a0, ka, lane);
    load_frag(a1, ka + 32, lane);
  }

  AF b0A, b1A, b0B, b1B;
  auto load_p = [&](AF& B0, AF& B1, int pt) {
    const _Float16* pc = pb + (size_t)(pt * 16 + lo) * Hc;
    load_frag(B0, pc, lane);
    load_frag(B1, pc + 32, lane);
  };
  auto emit = [&](AF& B0, AF& B1, int pt) {
    CF acc;
#pragma unroll
    for (int i = 0; i < 8; ++i) acc.f[i] = 0.f;
    wmma_acc(a0, B0, acc);
    wmma_acc(a1, B1, acc);
#pragma unroll
    for (int i = 0; i < 8; ++i) {
      const int row = r0 + i + hi8;
      p2c[((size_t)bh * Sc + row) * PT + pt * 16 + lo] = (_Float16)acc.f[i];
    }
  };

  load_p(b0A, b1A, 0);
  for (int pt = 0; pt < PT / 16; pt += 2) {
    load_p(b0B, b1B, pt + 1);
    emit(b0A, b1A, pt);
    if (pt + 2 < PT / 16) load_p(b0A, b1A, pt + 2);
    emit(b0B, b1B, pt + 1);
  }
}

// ---------------------------------------------------------------------------
// Fused flash attention with disentangled position bias.
// grid = (S/32, B*NH); block = 64 (2 waves); wave owns 16 q-rows.
// All dense loads of a k-tile (K frags AND V frags) are issued at the top of
// the iteration so their latency hides under the softmax VALU section.
// ---------------------------------------------------------------------------
__global__ __launch_bounds__(64) void k_attn(
    const _Float16* __restrict__ qh, const _Float16* __restrict__ kh,
    const _Float16* __restrict__ vt, const _Float16* __restrict__ posk,
    const _Float16* __restrict__ p2c, const int* __restrict__ relpos,
    const int* __restrict__ amask, _Float16* __restrict__ ctx)
{
  __shared__ __align__(16) _Float16 c2p_lds[32][PT];   // 32 KB
  __shared__ __align__(16) _Float16 p_lds[2][16][32];  // 2 KB

  const int lane = threadIdx.x & 31;
  const int wave = threadIdx.x >> 5;
  const int lo   = lane & 15;
  const int hi8  = (lane >> 4) << 3;
  const int bh   = blockIdx.y;
  const int bb   = bh / NHc;
  const int hh   = bh % NHc;
  const int q0   = blockIdx.x * 32 + wave * 16;

  const _Float16* qb  = qh + (size_t)bb * Sc * Hc + hh * DHc;
  const _Float16* kb  = kh + (size_t)bb * Sc * Hc + hh * DHc;
  const _Float16* vtb = vt + (size_t)bh * DHc * Sc;    // [DH][S], K-contig
  const _Float16* pk  = posk + hh * DHc;
  const int* mbase = amask + (size_t)bb * Sc * Sc;

  // Preload the wave's q-rows as two A fragments (d = 0..31, 32..63).
  AF aq0, aq1;
  {
    const _Float16* qa = qb + (size_t)(q0 + lo) * Hc;
    load_frag(aq0, qa, lane);
    load_frag(aq1, qa + 32, lane);
  }

  // Phase 1: c2p tile (16 q-rows x 512 positions) into LDS, double-buffered.
  {
    AF b0A, b1A, b0B, b1B;
    auto load_p = [&](AF& B0, AF& B1, int pt) {
      const _Float16* pc = pk + (size_t)(pt * 16 + lo) * Hc;
      load_frag(B0, pc, lane);
      load_frag(B1, pc + 32, lane);
    };
    auto emit = [&](AF& B0, AF& B1, int pt) {
      CF acc;
#pragma unroll
      for (int i = 0; i < 8; ++i) acc.f[i] = 0.f;
      wmma_acc(aq0, B0, acc);
      wmma_acc(aq1, B1, acc);
#pragma unroll
      for (int i = 0; i < 8; ++i)
        c2p_lds[wave * 16 + i + hi8][pt * 16 + lo] = (_Float16)acc.f[i];
    };
    load_p(b0A, b1A, 0);
    for (int pt = 0; pt < PT / 16; pt += 2) {
      load_p(b0B, b1B, pt + 1);
      emit(b0A, b1A, pt);
      if (pt + 2 < PT / 16) load_p(b0A, b1A, pt + 2);
      emit(b0B, b1B, pt + 1);
    }
  }
  __syncthreads();

  const float inv_scale = rsqrtf((float)(DHc * 3));   // 1/sqrt(DH*SCALE_FACTOR)
  float mrun[8], lrun[8];
  CF o[4];
#pragma unroll
  for (int i = 0; i < 8; ++i) { mrun[i] = -1e30f; lrun[i] = 0.f; }
#pragma unroll
  for (int t = 0; t < 4; ++t)
#pragma unroll
    for (int i = 0; i < 8; ++i) o[t].f[i] = 0.f;

  for (int kt = 0; kt < Sc / 32; ++kt) {
    const int kc0 = kt * 32;

    // ---- issue ALL dense loads for this tile up front ----
    AF bk0, bk1, bk2, bk3;        // K fragments (2 col-tiles x 2 d-halves)
    {
      const _Float16* kc0p = kb + (size_t)(kc0 + lo) * Hc;
      const _Float16* kc1p = kb + (size_t)(kc0 + 16 + lo) * Hc;
      load_frag(bk0, kc0p, lane);
      load_frag(bk1, kc0p + 32, lane);
      load_frag(bk2, kc1p, lane);
      load_frag(bk3, kc1p + 32, lane);
    }
    AF bvf[4];                    // V fragments for the later P@V WMMAs
#pragma unroll
    for (int dt = 0; dt < 4; ++dt)
      load_frag(bvf[dt], vtb + (size_t)(dt * 16 + lo) * Sc + kc0, lane);

    if (kt + 1 < Sc / 32) {
      __builtin_prefetch(kb + (size_t)(kc0 + 32 + lo) * Hc, 0, 1);
      __builtin_prefetch(vtb + (size_t)lo * Sc + kc0 + 32, 0, 1);
    }

    // --- content scores q.kT ---
    CF scf[2];
#pragma unroll
    for (int t = 0; t < 2; ++t)
#pragma unroll
      for (int i = 0; i < 8; ++i) scf[t].f[i] = 0.f;
    wmma_acc(aq0, bk0, scf[0]);
    wmma_acc(aq1, bk1, scf[0]);
    wmma_acc(aq0, bk2, scf[1]);
    wmma_acc(aq1, bk3, scf[1]);

    // --- add gathered c2p (LDS) and p2c (L2-resident global), apply mask ---
    float sv[2][8], pv[2][8];
#pragma unroll
    for (int t = 0; t < 2; ++t) {
#pragma unroll
      for (int i = 0; i < 8; ++i) {
        const int qrow = q0 + i + hi8;
        const int kcol = kc0 + t * 16 + lo;
        const int r = relpos[(size_t)qrow * Sc + kcol];
        int i1 = r + 256;  i1 = i1 < 0 ? 0 : (i1 > 511 ? 511 : i1);
        int i2 = 256 - r;  i2 = i2 < 0 ? 0 : (i2 > 511 ? 511 : i2);
        const float c2pv = (float)c2p_lds[wave * 16 + i + hi8][i1];
        const float p2cv = (float)p2c[((size_t)bh * Sc + kcol) * PT + i2];
        const int msk = mbase[(size_t)qrow * Sc + kcol];
        sv[t][i] = msk ? (scf[t].f[i] + c2pv + p2cv) * inv_scale : -1e30f;
      }
    }

    // --- online softmax (16-lane row reductions within each half-wave) ---
    float rmax[8];
#pragma unroll
    for (int i = 0; i < 8; ++i) rmax[i] = fmaxf(sv[0][i], sv[1][i]);
#pragma unroll
    for (int d = 1; d < 16; d <<= 1)
#pragma unroll
      for (int i = 0; i < 8; ++i)
        rmax[i] = fmaxf(rmax[i], __shfl_xor(rmax[i], d, 32));

    float alpha[8], rsum[8];
#pragma unroll
    for (int i = 0; i < 8; ++i) {
      const float mnew = fmaxf(mrun[i], rmax[i]);
      alpha[i] = __expf(mrun[i] - mnew);
      mrun[i] = mnew;
      const float p0 = (sv[0][i] > -1e29f) ? __expf(sv[0][i] - mnew) : 0.f;
      const float p1 = (sv[1][i] > -1e29f) ? __expf(sv[1][i] - mnew) : 0.f;
      pv[0][i] = p0; pv[1][i] = p1;
      rsum[i] = p0 + p1;
#pragma unroll
      for (int t = 0; t < 4; ++t) o[t].f[i] *= alpha[i];
    }
#pragma unroll
    for (int d = 1; d < 16; d <<= 1)
#pragma unroll
      for (int i = 0; i < 8; ++i) rsum[i] += __shfl_xor(rsum[i], d, 32);
#pragma unroll
    for (int i = 0; i < 8; ++i) lrun[i] = lrun[i] * alpha[i] + rsum[i];

    // --- stage P (D-layout -> A-layout) through per-wave LDS ---
#pragma unroll
    for (int t = 0; t < 2; ++t)
#pragma unroll
      for (int i = 0; i < 8; ++i)
        p_lds[wave][i + hi8][t * 16 + lo] = (_Float16)pv[t][i];
    __syncthreads();
    AF ap;
    load_frag(ap, &p_lds[wave][lo][0], lane);   // 2 x ds_load_b128
    __syncthreads();

    // --- O += P @ V  (V fragments were loaded at the top of the tile) ---
#pragma unroll
    for (int dt = 0; dt < 4; ++dt)
      wmma_acc(ap, bvf[dt], o[dt]);
  }

  float inv[8];
#pragma unroll
  for (int i = 0; i < 8; ++i) inv[i] = lrun[i] > 0.f ? 1.f / lrun[i] : 0.f;
#pragma unroll
  for (int dt = 0; dt < 4; ++dt)
#pragma unroll
    for (int i = 0; i < 8; ++i) {
      const int row = q0 + i + hi8;
      ctx[((size_t)bb * Sc + row) * Hc + hh * DHc + dt * 16 + lo] =
          (_Float16)(o[dt].f[i] * inv[i]);
    }
}

// ---------------------------------------------------------------------------
// Row LayerNorm: out = (x - mu) * rsqrt(var + eps) * g + b
// ---------------------------------------------------------------------------
__global__ __launch_bounds__(256) void k_lnorm(
    const float* __restrict__ x, const float* __restrict__ g,
    const float* __restrict__ beta, float* __restrict__ out, int N)
{
  __shared__ float red[8];
  const int row  = blockIdx.x;
  const float* xr = x + (size_t)row * N;
  const int lane = threadIdx.x & 31;
  const int wave = threadIdx.x >> 5;

  float s = 0.f;
  for (int j = threadIdx.x; j < N; j += 256) s += xr[j];
#pragma unroll
  for (int d = 1; d < 32; d <<= 1) s += __shfl_xor(s, d, 32);
  if (lane == 0) red[wave] = s;
  __syncthreads();
  if (threadIdx.x == 0) {
    float t = 0.f;
    for (int w = 0; w < 8; ++w) t += red[w];
    red[0] = t;
  }
  __syncthreads();
  const float mu = red[0] / (float)N;
  __syncthreads();

  float v = 0.f;
  for (int j = threadIdx.x; j < N; j += 256) {
    const float d = xr[j] - mu;
    v += d * d;
  }
#pragma unroll
  for (int d = 1; d < 32; d <<= 1) v += __shfl_xor(v, d, 32);
  if (lane == 0) red[wave] = v;
  __syncthreads();
  if (threadIdx.x == 0) {
    float t = 0.f;
    for (int w = 0; w < 8; ++w) t += red[w];
    red[0] = t;
  }
  __syncthreads();
  const float var = red[0] / (float)N;
  const float rs = rsqrtf(var + 1e-7f);

  for (int j = threadIdx.x; j < N; j += 256)
    out[(size_t)row * N + j] = (xr[j] - mu) * rs * g[j] + beta[j];
}

// ---------------------------------------------------------------------------
extern "C" void kernel_launch(void* const* d_in, const int* in_sizes, int n_in,
                              void* d_out, int out_size, void* d_ws, size_t ws_size,
                              hipStream_t stream) {
  (void)in_sizes; (void)n_in; (void)out_size; (void)ws_size;

  const float* hs     = (const float*)d_in[0];
  const float* rel    = (const float*)d_in[1];
  const float* Wq     = (const float*)d_in[2];
  const float* bq     = (const float*)d_in[3];
  const float* Wk     = (const float*)d_in[4];
  const float* bk     = (const float*)d_in[5];
  const float* Wv     = (const float*)d_in[6];
  const float* bvv    = (const float*)d_in[7];
  const float* Wo     = (const float*)d_in[8];
  const float* bo     = (const float*)d_in[9];
  const float* lng    = (const float*)d_in[10];
  const float* lnb    = (const float*)d_in[11];
  const int*   amask  = (const int*)d_in[12];
  const int*   relpos = (const int*)d_in[13];
  float* out = (float*)d_out;

  char* ws = (char*)d_ws;
  size_t off = 0;
  auto alloc = [&](size_t bytes) {
    char* p = ws + off;
    off += (bytes + 255) & ~(size_t)255;
    return p;
  };
  _Float16* hs16 = (_Float16*)alloc((size_t)Bc * Sc * Hc * sizeof(_Float16));
  _Float16* re16 = (_Float16*)alloc((size_t)PT * Hc * sizeof(_Float16));
  _Float16* Wqt  = (_Float16*)alloc((size_t)Hc * Hc * sizeof(_Float16));
  _Float16* Wkt  = (_Float16*)alloc((size_t)Hc * Hc * sizeof(_Float16));
  _Float16* Wvt  = (_Float16*)alloc((size_t)Hc * Hc * sizeof(_Float16));
  _Float16* Wot  = (_Float16*)alloc((size_t)Hc * Hc * sizeof(_Float16));
  _Float16* qh   = (_Float16*)alloc((size_t)Bc * Sc * Hc * sizeof(_Float16));
  _Float16* khp  = (_Float16*)alloc((size_t)Bc * Sc * Hc * sizeof(_Float16));
  _Float16* vt   = (_Float16*)alloc((size_t)Bc * Sc * Hc * sizeof(_Float16));
  _Float16* posk = (_Float16*)alloc((size_t)PT * Hc * sizeof(_Float16));
  _Float16* posq = (_Float16*)alloc((size_t)PT * Hc * sizeof(_Float16));
  _Float16* p2cb = (_Float16*)alloc((size_t)Bc * NHc * Sc * PT * sizeof(_Float16));
  _Float16* ctxb = (_Float16*)alloc((size_t)Bc * Sc * Hc * sizeof(_Float16));
  float*    tmp  = (float*)   alloc((size_t)Bc * Sc * Hc * sizeof(float));

  const int MS = Bc * Sc;                       // 4096 token rows

  // f32 -> f16 activation copies.
  k_cvt16<<<(MS * Hc / 4 + 255) / 256, 256, 0, stream>>>(hs, hs16, MS * Hc);
  k_cvt16<<<(PT * Hc / 4 + 255) / 256, 256, 0, stream>>>(rel, re16, PT * Hc);

  // Weight transposes (f16, K-contiguous B operands).
  dim3 gwt(Hc / 32, Hc / 32);
  k_wT<<<gwt, 128, 0, stream>>>(Wq, Wqt, Hc, Hc);
  k_wT<<<gwt, 128, 0, stream>>>(Wk, Wkt, Hc, Hc);
  k_wT<<<gwt, 128, 0, stream>>>(Wv, Wvt, Hc, Hc);
  k_wT<<<gwt, 128, 0, stream>>>(Wo, Wot, Hc, Hc);

  // Projections.
  dim3 gproj(Hc / 64, MS / 128);                // (24, 32)
  dim3 gpos (Hc / 64, PT / 128);                // (24, 4)
  k_gemm16   <<<gproj, 128, 0, stream>>>(hs16, Wqt, bq,  qh,   MS, Hc, Hc);
  k_gemm16   <<<gproj, 128, 0, stream>>>(hs16, Wkt, bk,  khp,  MS, Hc, Hc);
  k_gemm16_vT<<<gproj, 128, 0, stream>>>(hs16, Wvt, bvv, vt,   MS, Hc, Hc);
  k_gemm16   <<<gpos,  128, 0, stream>>>(re16, Wkt, bk,  posk, PT, Hc, Hc);
  k_gemm16   <<<gpos,  128, 0, stream>>>(re16, Wqt, bq,  posq, PT, Hc, Hc);

  // p2c_full (L2-resident gather table for the flash loop).
  dim3 gp2c(Sc / 64, Bc * NHc);                 // (16, 96)
  k_p2c<<<gp2c, 128, 0, stream>>>(khp, posq, p2cb);

  // Fused flash attention.
  dim3 gattn(Sc / 32, Bc * NHc);                // (32, 96)
  k_attn<<<gattn, 64, 0, stream>>>(qh, khp, vt, posk, p2cb, relpos, amask, ctxb);

  // Output projection + residual, then LayerNorm.
  k_gemm16_out<<<gproj, 128, 0, stream>>>(ctxb, Wot, bo, hs, tmp, MS, Hc, Hc);
  k_lnorm<<<MS, 256, 0, stream>>>(tmp, lng, lnb, out, Hc);
}
